// BandScore_28784870818045
// MI455X (gfx1250) — matrix operand order repaired and checked by reading, not compiled, not run on hardware
//
#include <hip/hip_runtime.h>
#include <hip/hip_bf16.h>

// ---------------------------------------------------------------------------
// Restormer-style block for MI455X (gfx1250, wave32).
// All 1x1-conv GEMMs + attention GEMMs run on v_wmma_f32_16x16x32_bf16
// (bf16 in / f32 accumulate). Streaming ops (LN, dw3x3, GELU, pools) are
// bandwidth-bound; activations fit in the 192MB L2 so multi-pass reads re-hit L2.
// f32->bf16 conversions use the hardware fptrunc path (single VALU op) instead
// of a manual RNE bit sequence.
// ---------------------------------------------------------------------------

#define DIMC 128
#define NHEADS 2
#define HD 64
#define BB 4
#define HH 192
#define WW 192
#define HWSZ (HH * WW) /* 36864 */
#define LNEPS 1e-5f

typedef __attribute__((ext_vector_type(16))) __bf16 v16bf;
typedef __attribute__((ext_vector_type(8)))  float  v8f;

union Frag { v16bf v; unsigned short u[16]; };

__device__ __forceinline__ unsigned short f2bf(float f) {
  __bf16 h = (__bf16)f;               // hardware RNE f32->bf16
  return __builtin_bit_cast(unsigned short, h);
}
__device__ __forceinline__ float geluf(float x) {
  return 0.5f * x * (1.0f + erff(x * 0.70710678118654752440f));
}

// -------------------------------------------------- weight f32 -> bf16 prep
__global__ void k_f32_to_bf16(const float* __restrict__ src,
                              unsigned short* __restrict__ dst, int n) {
  int i = blockIdx.x * 256 + threadIdx.x;
  if (i < n) dst[i] = f2bf(src[i]);
}

// -------------------------------------------------- channel LayerNorm -> bf16
// one thread per spatial position; coalesced across positions for each channel
__global__ void k_ln_c(const float* __restrict__ x, const float* __restrict__ w,
                       const float* __restrict__ bia,
                       unsigned short* __restrict__ out_bf) {
  int p = blockIdx.x * 256 + threadIdx.x;  // [0, BB*HWSZ)
  int b = p / HWSZ, hw = p % HWSZ;
  const float* xp = x + (size_t)b * DIMC * HWSZ + hw;
  float s = 0.f, ss = 0.f;
  for (int c = 0; c < DIMC; ++c) {
    float v = xp[(size_t)c * HWSZ];
    s += v; ss += v * v;
  }
  float mu   = s  * (1.0f / DIMC);
  float var  = ss * (1.0f / DIMC) - mu * mu;      // biased variance
  float rstd = rsqrtf(var + LNEPS);
  unsigned short* op = out_bf + (size_t)b * DIMC * HWSZ + hw;
  for (int c = 0; c < DIMC; ++c) {
    float v = (xp[(size_t)c * HWSZ] - mu) * rstd * w[c] + bia[c];
    op[(size_t)c * HWSZ] = f2bf(v);
  }
}

// -------------------------------------------------- pointwise conv as GEMM
// Y[b,co,p] = act( sum_ci Wbf[co,ci] * Xbf[b,ci,p] ); Co=128, CI=128 or 256.
// Block: 256 thr (8 waves). Block tile M=128 x N=64; wave tile 16 x 64.
template <int CI, int DOGELU>
__global__ void k_gemm_pw(const unsigned short* __restrict__ Wbf,
                          const unsigned short* __restrict__ Xbf,
                          float* __restrict__ Y) {
  __shared__ unsigned short Bs[32][72];  // 32(K) x 64(N) bf16, row pad to 144B
  const int tid  = threadIdx.x;
  const int lane = tid & 31, wv = tid >> 5;
  const int half = lane >> 4;
  const int b    = blockIdx.y;
  const int n0   = blockIdx.x * 64;
  const int m0   = wv * 16;
  const int rowA = m0 + (lane & 15);

  v8f acc[4];
  for (int t = 0; t < 4; ++t)
    for (int r = 0; r < 8; ++r) acc[t][r] = 0.0f;

  const int ldRow = tid >> 3;        // 0..31  (K row)
  const int ldCol = (tid & 7) * 8;   // 0..56  (N col, 8 bf16 = 16B)
  const unsigned short* xbase = Xbf + (size_t)b * CI * HWSZ + n0 + ldCol;

  for (int k0 = 0; k0 < CI; k0 += 32) {
    // stage the shared B tile (K x N) through LDS
    uint4 bt = *(const uint4*)(xbase + (size_t)(k0 + ldRow) * HWSZ);
    *(uint4*)&Bs[ldRow][ldCol] = bt;
    if (k0 + 32 < CI)  // lookahead -> global_prefetch_b8
      __builtin_prefetch(xbase + (size_t)(k0 + 32 + ldRow) * HWSZ, 0, 1);
    __syncthreads();

    // A fragment: 16x32 bf16, contiguous k = 16*half + s (same order as B)
    Frag af;
    const unsigned short* wrow = Wbf + rowA * CI + k0;
    for (int v2 = 0; v2 < 8; ++v2) {
      int kk = (half << 4) + (v2 << 1);
      unsigned pr = *(const unsigned*)(wrow + kk);
      af.u[2 * v2]     = (unsigned short)(pr & 0xFFFFu);
      af.u[2 * v2 + 1] = (unsigned short)(pr >> 16);
    }
    for (int t = 0; t < 4; ++t) {
      Frag bfr;
      int col = t * 16 + (lane & 15);
      for (int e = 0; e < 16; ++e) bfr.u[e] = Bs[half * 16 + e][col];
      acc[t] = __builtin_amdgcn_wmma_f32_16x16x32_bf16(
          false, af.v, false, bfr.v, (short)0, acc[t], false, false);
    }
    __syncthreads();
  }

  for (int t = 0; t < 4; ++t) {
    int col = n0 + t * 16 + (lane & 15);
    for (int r = 0; r < 8; ++r) {
      int m = m0 + r + half * 8;
      float vv = acc[t][r];
      if (DOGELU) vv = geluf(vv);
      Y[((size_t)b * DIMC + m) * HWSZ + col] = vv;
    }
  }
}

// -------------------------------------------------- depthwise 3x3, pad 1
__global__ void k_dwconv3(const float* __restrict__ in,
                          const float* __restrict__ wdw,
                          float* __restrict__ out) {
  const int bc = blockIdx.y;                       // b*DIMC + c
  const int c  = bc & (DIMC - 1);
  const int p  = blockIdx.x * 256 + threadIdx.x;   // pixel
  const int y  = p / WW, x = p % WW;
  const float* ip = in + (size_t)bc * HWSZ;
  const float* wp = wdw + c * 9;
  float s = 0.f;
  for (int dy = -1; dy <= 1; ++dy) {
    int yy = y + dy;
    if (yy < 0 || yy >= HH) continue;
    for (int dx = -1; dx <= 1; ++dx) {
      int xx = x + dx;
      if (xx < 0 || xx >= WW) continue;
      s += wp[(dy + 1) * 3 + (dx + 1)] * ip[yy * WW + xx];
    }
  }
  out[(size_t)bc * HWSZ + p] = s;
}

// -------------------------------------------------- L2 norm over HW, in place
__global__ void k_l2norm(float* __restrict__ q) {
  __shared__ float red[8];
  const int row = blockIdx.x;  // b*DIMC + c  (512 rows)
  float* rp = q + (size_t)row * HWSZ;
  float ss = 0.f;
  for (int i = threadIdx.x; i < HWSZ; i += 256) { float v = rp[i]; ss += v * v; }
  for (int o = 16; o; o >>= 1) ss += __shfl_xor(ss, o, 32);
  if ((threadIdx.x & 31) == 0) red[threadIdx.x >> 5] = ss;
  __syncthreads();
  float tot = red[0] + red[1] + red[2] + red[3] + red[4] + red[5] + red[6] + red[7];
  float inv = 1.0f / fmaxf(sqrtf(tot), 1e-12f);
  for (int i = threadIdx.x; i < HWSZ; i += 256) rp[i] *= inv;
}

__global__ void k_zero(float* p, int n) {
  int i = blockIdx.x * 256 + threadIdx.x;
  if (i < n) p[i] = 0.f;
}

// -------------------------------------------------- attn = q . k^T (split-K WMMA)
// grid (8 K-chunks, B*N); each block accumulates a partial 64x64 via f32 atomics
__global__ void k_attn_qk(const float* __restrict__ q, const float* __restrict__ k,
                          float* __restrict__ att) {
  const int lane = threadIdx.x & 31, wv = threadIdx.x >> 5;
  const int half = lane >> 4;
  const int bn = blockIdx.y, bsel = bn >> 1, n = bn & 1;
  const int L = HWSZ / 8;
  const int hw0 = blockIdx.x * L;
  const int mt = wv >> 1;
  const int ntb = (wv & 1) * 2;
  const float* qb = q + ((size_t)bsel * DIMC + n * HD) * HWSZ;
  const float* kb = k + ((size_t)bsel * DIMC + n * HD) * HWSZ;

  v8f acc[2];
  for (int t = 0; t < 2; ++t)
    for (int r = 0; r < 8; ++r) acc[t][r] = 0.f;

  const int rowA = mt * 16 + (lane & 15);
  const float* qrow = qb + (size_t)rowA * HWSZ + hw0;

  for (int kk = 0; kk < L; kk += 32) {
    Frag af;
    for (int v2 = 0; v2 < 8; ++v2) {
      int kidx = kk + (half << 4) + (v2 << 1);
      float2 pr = *(const float2*)(qrow + kidx);
      af.u[2 * v2]     = f2bf(pr.x);
      af.u[2 * v2 + 1] = f2bf(pr.y);
    }
    for (int t = 0; t < 2; ++t) {
      int d = (ntb + t) * 16 + (lane & 15);
      const float* krow = kb + (size_t)d * HWSZ + hw0 + kk + half * 16;
      Frag bfr;
      for (int e = 0; e < 16; ++e) bfr.u[e] = f2bf(krow[e]);
      acc[t] = __builtin_amdgcn_wmma_f32_16x16x32_bf16(
          false, af.v, false, bfr.v, (short)0, acc[t], false, false);
    }
  }
  for (int t = 0; t < 2; ++t) {
    int d = (ntb + t) * 16 + (lane & 15);
    for (int r = 0; r < 8; ++r) {
      int m = mt * 16 + r + half * 8;
      atomicAdd(&att[((size_t)bn * HD + m) * HD + d], acc[t][r]);  // f32 atomic
    }
  }
}

// ---------------------------------- scale, emit attn_weight, row softmax(64)
__global__ void k_softmax(const float* __restrict__ att, const float* __restrict__ scale,
                          float* __restrict__ attw_out, float* __restrict__ a) {
  const int lane = threadIdx.x & 31, wv = threadIdx.x >> 5;
  const int row = blockIdx.x * 8 + wv;  // 512 rows (b,n,c)
  const int n = (row >> 6) & 1;
  float s = scale[n];
  float v0 = att[row * 64 + lane] * s;
  float v1 = att[row * 64 + 32 + lane] * s;
  attw_out[row * 64 + lane] = v0;       // pre-softmax attn_weight output
  attw_out[row * 64 + 32 + lane] = v1;
  float mx = fmaxf(v0, v1);
  for (int o = 16; o; o >>= 1) mx = fmaxf(mx, __shfl_xor(mx, o, 32));
  float e0 = __expf(v0 - mx), e1 = __expf(v1 - mx);
  float sm = e0 + e1;
  for (int o = 16; o; o >>= 1) sm += __shfl_xor(sm, o, 32);
  float inv = 1.0f / sm;
  a[row * 64 + lane] = e0 * inv;
  a[row * 64 + 32 + lane] = e1 * inv;
}

// -------------------------------------------------- out = a . v (WMMA)
__global__ void k_av(const float* __restrict__ a, const float* __restrict__ v,
                     float* __restrict__ out) {
  __shared__ unsigned short Vs[32][72];
  const int tid = threadIdx.x, lane = tid & 31, wv_ = tid >> 5, half = lane >> 4;
  const int bn = blockIdx.y, bsel = bn >> 1, n = bn & 1;
  const int hw0 = blockIdx.x * 64;
  const int mt = wv_ >> 1, ntb = (wv_ & 1) * 2;
  const float* ab = a + (size_t)bn * HD * HD;
  const float* vb = v + ((size_t)bsel * DIMC + n * HD) * HWSZ;

  v8f acc[2];
  for (int t = 0; t < 2; ++t)
    for (int r = 0; r < 8; ++r) acc[t][r] = 0.f;

  const int ldRow = tid >> 3, ldCol = (tid & 7) * 8;
  const int rowA = mt * 16 + (lane & 15);

  for (int ks = 0; ks < 2; ++ks) {
    const float* vp = vb + (size_t)(ks * 32 + ldRow) * HWSZ + hw0 + ldCol;
    for (int j = 0; j < 8; ++j) Vs[ldRow][ldCol + j] = f2bf(vp[j]);
    __syncthreads();
    Frag af;
    const float* arow = ab + rowA * HD + ks * 32;
    for (int v2 = 0; v2 < 8; ++v2) {
      int kidx = (half << 4) + (v2 << 1);
      float2 pr = *(const float2*)(arow + kidx);
      af.u[2 * v2]     = f2bf(pr.x);
      af.u[2 * v2 + 1] = f2bf(pr.y);
    }
    for (int t = 0; t < 2; ++t) {
      Frag bfr;
      int col = (ntb + t) * 16 + (lane & 15);
      for (int e = 0; e < 16; ++e) bfr.u[e] = Vs[half * 16 + e][col];
      acc[t] = __builtin_amdgcn_wmma_f32_16x16x32_bf16(
          false, af.v, false, bfr.v, (short)0, acc[t], false, false);
    }
    __syncthreads();
  }
  for (int t = 0; t < 2; ++t) {
    int col = hw0 + (ntb + t) * 16 + (lane & 15);
    for (int r = 0; r < 8; ++r) {
      int m = mt * 16 + r + half * 8;
      out[((size_t)bsel * DIMC + n * HD + m) * HWSZ + col] = acc[t][r];
    }
  }
}

// -------------------------------------------------- gelu(cat(o1,o2)) -> bf16
__global__ void k_gelucat(const float* __restrict__ o1, const float* __restrict__ o2,
                          unsigned short* __restrict__ cat) {
  size_t i = (size_t)blockIdx.x * 256 + threadIdx.x;  // BB*2*DIMC*HWSZ
  int hw = (int)(i % HWSZ);
  int ch = (int)((i / HWSZ) % (2 * DIMC));
  int b  = (int)(i / ((size_t)2 * DIMC * HWSZ));
  float s = (ch < DIMC) ? o1[((size_t)b * DIMC + ch) * HWSZ + hw]
                        : o2[((size_t)b * DIMC + (ch - DIMC)) * HWSZ + hw];
  cat[i] = f2bf(geluf(s));
}

// -------------------------------------------------- avg / max pool over HW
__global__ void k_pool(const float* __restrict__ f, float* __restrict__ avg,
                       float* __restrict__ mx) {
  __shared__ float rs[8], rm[8];
  int row = blockIdx.x;  // b*DIMC + c
  const float* rp = f + (size_t)row * HWSZ;
  float s = 0.f, m = -3.4e38f;
  for (int i = threadIdx.x; i < HWSZ; i += 256) {
    float v = rp[i]; s += v; m = fmaxf(m, v);
  }
  for (int o = 16; o; o >>= 1) {
    s += __shfl_xor(s, o, 32);
    m = fmaxf(m, __shfl_xor(m, o, 32));
  }
  if ((threadIdx.x & 31) == 0) { rs[threadIdx.x >> 5] = s; rm[threadIdx.x >> 5] = m; }
  __syncthreads();
  if (threadIdx.x == 0) {
    float S = 0.f, M = -3.4e38f;
    for (int i = 0; i < 8; ++i) { S += rs[i]; M = fmaxf(M, rm[i]); }
    avg[row] = S / (float)HWSZ;
    mx[row] = M;
  }
}

// -------------------------------------------------- channel gate MLP
__global__ void k_gate(const float* __restrict__ avg, const float* __restrict__ mx,
                       const float* __restrict__ w1, const float* __restrict__ b1,
                       const float* __restrict__ w2, const float* __restrict__ b2,
                       float* __restrict__ outp) {
  __shared__ float hid[16];
  int b = blockIdx.x, t = threadIdx.x;
  if (t < 16) {
    int j = t & 7;
    const float* p = (t < 8) ? (avg + b * DIMC) : (mx + b * DIMC);
    float s = b1[j];
    for (int c = 0; c < DIMC; ++c) s += w1[j * DIMC + c] * p[c];
    hid[t] = fmaxf(s, 0.f);
  }
  __syncthreads();
  float sa = b2[t], sm = b2[t];  // b2 added in each mlp() (matches reference)
  for (int j = 0; j < 8; ++j) {
    sa += w2[t * 8 + j] * hid[j];
    sm += w2[t * 8 + j] * hid[8 + j];
  }
  float z = sa + sm;
  outp[b * DIMC + t] = 1.0f / (1.0f + __expf(-z));
}

// ===========================================================================
extern "C" void kernel_launch(void* const* d_in, const int* in_sizes, int n_in,
                              void* d_out, int out_size, void* d_ws, size_t ws_size,
                              hipStream_t stream) {
  const float* x       = (const float*)d_in[0];
  const float* lnin_w  = (const float*)d_in[1];
  const float* lnin_b  = (const float*)d_in[2];
  const float* wq_pw   = (const float*)d_in[3];
  const float* wq_dw   = (const float*)d_in[4];
  const float* wk_pw   = (const float*)d_in[5];
  const float* wk_dw   = (const float*)d_in[6];
  const float* wv_pw   = (const float*)d_in[7];
  const float* wv_dw   = (const float*)d_in[8];
  const float* scale   = (const float*)d_in[9];
  const float* lnout_w = (const float*)d_in[10];
  const float* lnout_b = (const float*)d_in[11];
  const float* f1_pw   = (const float*)d_in[12];
  const float* f1_dw   = (const float*)d_in[13];
  const float* f2_pw   = (const float*)d_in[14];
  const float* f2_dw   = (const float*)d_in[15];
  const float* f_out   = (const float*)d_in[16];
  const float* g_w1    = (const float*)d_in[17];
  const float* g_b1    = (const float*)d_in[18];
  const float* g_w2    = (const float*)d_in[19];
  const float* g_b2    = (const float*)d_in[20];
  float* outp = (float*)d_out;

  // ---- workspace layout (needs ~340 MB) ----
  const size_t S_EL = (size_t)BB * DIMC * HWSZ;   // 18,874,368 elems
  char* ws = (char*)d_ws;
  unsigned short* WBF = (unsigned short*)ws;                    // 229 KB bf16 weights
  float* ATT  = (float*)(ws + 240 * 1024);                      // 128 KB
  float* ABUF = (float*)(ws + 240 * 1024 + 128 * 1024);         // 128 KB
  float* PAVG = (float*)(ws + 240 * 1024 + 256 * 1024);
  float* PMAX = PAVG + BB * DIMC;
  size_t off = 512 * 1024;
  unsigned short* XBF = (unsigned short*)(ws + off); off += S_EL * 2;  // bf16 LN out
  float* T1 = (float*)(ws + off); off += S_EL * 4;   // pw temp / attn-out / cat(bf16)
  float* Q  = (float*)(ws + off); off += S_EL * 4;
  float* K  = (float*)(ws + off); off += S_EL * 4;
  float* V  = (float*)(ws + off);

  // weight prep: f32 -> bf16 (row-major co x ci)
  k_f32_to_bf16<<<64, 256, 0, stream>>>(wq_pw, WBF + 0,     16384);
  k_f32_to_bf16<<<64, 256, 0, stream>>>(wk_pw, WBF + 16384, 16384);
  k_f32_to_bf16<<<64, 256, 0, stream>>>(wv_pw, WBF + 32768, 16384);
  k_f32_to_bf16<<<64, 256, 0, stream>>>(f1_pw, WBF + 49152, 16384);
  k_f32_to_bf16<<<64, 256, 0, stream>>>(f2_pw, WBF + 65536, 16384);
  k_f32_to_bf16<<<128, 256, 0, stream>>>(f_out, WBF + 81920, 32768);

  // LN(x) -> bf16
  k_ln_c<<<BB * HWSZ / 256, 256, 0, stream>>>(x, lnin_w, lnin_b, XBF);

  dim3 gg(HWSZ / 64, BB);          // GEMM grid
  dim3 gd(HWSZ / 256, BB * DIMC);  // dwconv grid

  // q / k / v projections: WMMA GEMM -> dw3x3
  k_gemm_pw<128, 0><<<gg, 256, 0, stream>>>(WBF + 0, XBF, T1);
  k_dwconv3<<<gd, 256, 0, stream>>>(T1, wq_dw, Q);
  k_gemm_pw<128, 0><<<gg, 256, 0, stream>>>(WBF + 16384, XBF, T1);
  k_dwconv3<<<gd, 256, 0, stream>>>(T1, wk_dw, K);
  k_gemm_pw<128, 0><<<gg, 256, 0, stream>>>(WBF + 32768, XBF, T1);
  k_dwconv3<<<gd, 256, 0, stream>>>(T1, wv_dw, V);

  // L2 normalize q,k along HW
  k_l2norm<<<BB * DIMC, 256, 0, stream>>>(Q);
  k_l2norm<<<BB * DIMC, 256, 0, stream>>>(K);

  // attention logits (split-K WMMA + f32 atomics), scale + softmax
  k_zero<<<128, 256, 0, stream>>>(ATT, BB * NHEADS * HD * HD);
  dim3 ga(8, BB * NHEADS);
  k_attn_qk<<<ga, 256, 0, stream>>>(Q, K, ATT);
  k_softmax<<<64, 256, 0, stream>>>(ATT, scale, outp + BB * DIMC, ABUF);

  // out = a @ v
  dim3 gv(HWSZ / 64, BB * NHEADS);
  k_av<<<gv, 256, 0, stream>>>(ABUF, V, T1);

  // LN(out) -> bf16, FFN branches (GEMM+GELU epilogue -> dw3x3)
  k_ln_c<<<BB * HWSZ / 256, 256, 0, stream>>>(T1, lnout_w, lnout_b, XBF);
  k_gemm_pw<128, 1><<<gg, 256, 0, stream>>>(WBF + 49152, XBF, Q);
  k_dwconv3<<<gd, 256, 0, stream>>>(Q, f1_dw, K);   // o1
  k_gemm_pw<128, 1><<<gg, 256, 0, stream>>>(WBF + 65536, XBF, Q);
  k_dwconv3<<<gd, 256, 0, stream>>>(Q, f2_dw, V);   // o2

  // gelu(cat(o1,o2)) -> bf16 (reuses T1), then 256->128 WMMA GEMM
  k_gelucat<<<(unsigned)(S_EL * 2 / 256), 256, 0, stream>>>(K, V, (unsigned short*)T1);
  k_gemm_pw<256, 0><<<gg, 256, 0, stream>>>(WBF + 81920, (unsigned short*)T1, Q);

  // channel gate
  k_pool<<<BB * DIMC, 256, 0, stream>>>(Q, PAVG, PMAX);
  k_gate<<<BB, 128, 0, stream>>>(PAVG, PMAX, g_w1, g_b1, g_w2, g_b2, outp);

  (void)in_sizes; (void)n_in; (void)out_size; (void)ws_size;
}
